// AutoEncoderTopK_73212012528144
// MI455X (gfx1250) — matrix operand order repaired and checked by reading, not compiled, not run on hardware
//
#include <hip/hip_runtime.h>
#include <hip/hip_bf16.h>

typedef __attribute__((ext_vector_type(16))) __bf16 v16bf;
typedef __attribute__((ext_vector_type(8)))  __bf16 v8bf;
typedef __attribute__((ext_vector_type(8)))  float  v8f;

constexpr int N_TOK  = 16384;
constexpr int D_IN   = 768;
constexpr int D_SAE  = 24576;
constexpr int TOPK   = 64;

constexpr int TILE_M = 64;                 // tokens per workgroup (4 x 16 sub-tiles)
constexpr int SUBT   = 4;                  // sub-tiles sharing each B fragment
constexpr int CHUNK  = 128;                // features per chunk (8 waves x 16)
constexpr int NCHUNK = D_SAE / CHUNK;      // 192
constexpr int KBLK   = D_IN / 32;          // 24 k-blocks of 32
constexpr int NFRAG  = (D_SAE / 16) * KBLK; // 36864 B fragments

// ---------------------------------------------------------------------------
// Pre-pass: convert W_enc fp32 -> bf16 in WMMA-B-fragment order.
// Fragment (jt, kbi) = features [jt*16, +16) x K [kbi*32, +32); one wave each.
// Lane l supplies B[k][n] with n = l%16, k = 16*(l/16) + i  (i = 0..15),
// stored contiguously: Wq[frag*512 + l*16 + i].
// ---------------------------------------------------------------------------
__global__ __launch_bounds__(256)
void convert_W(const float* __restrict__ W, __bf16* __restrict__ Wq) {
  const int frag = blockIdx.x * 8 + (threadIdx.x >> 5);
  const int lane = threadIdx.x & 31;
  const int jt   = frag / KBLK;
  const int kbi  = frag - jt * KBLK;
  const float* src =
      &W[(size_t)(jt * 16 + (lane & 15)) * D_IN + kbi * 32 + (lane >> 4) * 16];
  v16bf b;
#pragma unroll
  for (int i = 0; i < 16; ++i) b[i] = (__bf16)src[i];
  *(v16bf*)&Wq[((size_t)frag << 9) + (lane << 4)] = b;
}

// ---------------------------------------------------------------------------
// Fused encoder GEMM (WMMA bf16) + exact per-token top-64 + sparse decode.
// PREW = true : B fragments loaded pre-swizzled bf16 from workspace.
// PREW = false: B fragments converted from fp32 W on the fly (fallback).
// ---------------------------------------------------------------------------
template <bool PREW>
__global__ __launch_bounds__(256)
void sae_topk_fused(const float* __restrict__ x,      // [N, D_IN]
                    const float* __restrict__ W,      // [D_SAE, D_IN] fp32
                    const __bf16* __restrict__ Wq,    // swizzled bf16 (PREW)
                    const float* __restrict__ benc,   // [D_SAE]
                    const float* __restrict__ dec,    // [D_SAE, D_IN]
                    const float* __restrict__ bdec,   // [D_IN]
                    float* __restrict__ out)          // [N, D_IN]
{
  __shared__ __bf16 xs[TILE_M * D_IN];     // 96 KB  bf16 (x - b_dec) tile
  __shared__ float  fs[TILE_M * CHUNK];    // 32 KB  f chunk
  __shared__ float  hv[TILE_M * TOPK];     // 16 KB  heap values
  __shared__ int    hidx[TILE_M * TOPK];   // 16 KB  heap indices

  const int tid   = threadIdx.x;
  const int lane  = tid & 31;
  const int wave  = tid >> 5;
  const int ncol  = lane & 15;   // N column within the 16x16 tile
  const int hfsel = lane >> 4;   // half-wave select (K striping)
  const int tokBase = blockIdx.x * TILE_M;

  // ---- stage x tile to LDS as bf16: (x - b_dec) ----
  for (int i = tid; i < TILE_M * D_IN; i += 256) {
    int t = i / D_IN, d = i - t * D_IN;
    xs[i] = (__bf16)(x[(size_t)(tokBase + t) * D_IN + d] - bdec[d]);
  }
  // ---- init per-token min-heaps (ReLU floor = 0) ----
  for (int i = tid; i < TILE_M * TOPK; i += 256) { hv[i] = 0.0f; hidx[i] = 0; }
  __syncthreads();

  // ---- loop over dictionary chunks ----
  for (int c = 0; c < NCHUNK; ++c) {
    const int j0 = c * CHUNK + wave * 16;         // this wave's 16 features
    const int jt = j0 >> 4;                       // feature-tile index

    // speculative prefetch of next chunk's B data -> global_prefetch_b8
    if (c + 1 < NCHUNK) {
      if constexpr (PREW) {
        const size_t nb = ((size_t)((c + 1) * (CHUNK / 16)) * KBLK) << 9;
        __builtin_prefetch(&Wq[nb + (size_t)tid * 768], 0, 1);
      } else {
        const int pj = (c + 1) * CHUNK + (tid >> 1);
        __builtin_prefetch(&W[(size_t)pj * D_IN + (size_t)(tid & 1) * (D_IN / 2)], 0, 1);
      }
    }

    v8f acc0 = {}, acc1 = {}, acc2 = {}, acc3 = {};
    const float*  wrow = &W[(size_t)(j0 + ncol) * D_IN + hfsel * 16];
    const __bf16* wqf  = &Wq[(((size_t)jt * KBLK) << 9) + (lane << 4)];
    const __bf16* xrow0 = &xs[( 0 + (lane & 15)) * D_IN];
    const __bf16* xrow1 = &xs[(16 + (lane & 15)) * D_IN];
    const __bf16* xrow2 = &xs[(32 + (lane & 15)) * D_IN];
    const __bf16* xrow3 = &xs[(48 + (lane & 15)) * D_IN];

#pragma unroll 2
    for (int kb = 0; kb < D_IN; kb += 32) {
      // B fragment (32x16 bf16, N = lane%16), shared by 4 token sub-tiles
      v16bf b;
      if constexpr (PREW) {
        b = *(const v16bf*)&wqf[(size_t)(kb >> 5) << 9];   // 32B/lane, coalesced
      } else {
#pragma unroll
        for (int i = 0; i < 16; ++i) b[i] = (__bf16)wrow[kb + i];
      }

      // A fragments (16x32 bf16, M = lane%16): k = 16*hfsel + i, matching B
      const int o0 = kb + 16 * hfsel;
      const int o1 = kb + 16 * hfsel + 8;
      v16bf a;
      v8bf lo, hi;

      lo = *(const v8bf*)&xrow0[o0]; hi = *(const v8bf*)&xrow0[o1];
#pragma unroll
      for (int i = 0; i < 8; ++i) { a[i] = lo[i]; a[i + 8] = hi[i]; }
      acc0 = __builtin_amdgcn_wmma_f32_16x16x32_bf16(false, a, false, b,
                                                     (short)0, acc0, false, false);

      lo = *(const v8bf*)&xrow1[o0]; hi = *(const v8bf*)&xrow1[o1];
#pragma unroll
      for (int i = 0; i < 8; ++i) { a[i] = lo[i]; a[i + 8] = hi[i]; }
      acc1 = __builtin_amdgcn_wmma_f32_16x16x32_bf16(false, a, false, b,
                                                     (short)0, acc1, false, false);

      lo = *(const v8bf*)&xrow2[o0]; hi = *(const v8bf*)&xrow2[o1];
#pragma unroll
      for (int i = 0; i < 8; ++i) { a[i] = lo[i]; a[i + 8] = hi[i]; }
      acc2 = __builtin_amdgcn_wmma_f32_16x16x32_bf16(false, a, false, b,
                                                     (short)0, acc2, false, false);

      lo = *(const v8bf*)&xrow3[o0]; hi = *(const v8bf*)&xrow3[o1];
#pragma unroll
      for (int i = 0; i < 8; ++i) { a[i] = lo[i]; a[i + 8] = hi[i]; }
      acc3 = __builtin_amdgcn_wmma_f32_16x16x32_bf16(false, a, false, b,
                                                     (short)0, acc3, false, false);
    }

    // ReLU + b_enc; C/D layout: VGPR r holds (m = r + 8*hfsel, col ncol)
    const float be = benc[j0 + ncol];
#pragma unroll
    for (int r = 0; r < 8; ++r) {
      const int mrow = r + 8 * hfsel;
      const int fcol = wave * 16 + ncol;
      float v0 = acc0[r] + be;
      float v1 = acc1[r] + be;
      float v2 = acc2[r] + be;
      float v3 = acc3[r] + be;
      fs[( 0 + mrow) * CHUNK + fcol] = v0 > 0.0f ? v0 : 0.0f;
      fs[(16 + mrow) * CHUNK + fcol] = v1 > 0.0f ? v1 : 0.0f;
      fs[(32 + mrow) * CHUNK + fcol] = v2 > 0.0f ? v2 : 0.0f;
      fs[(48 + mrow) * CHUNK + fcol] = v3 > 0.0f ? v3 : 0.0f;
    }
    __syncthreads();

    // one thread per token (64 in parallel): exact top-64 via LDS min-heap
    if (tid < TILE_M) {
      float* h = &hv[tid * TOPK];
      int*   g = &hidx[tid * TOPK];
      const float* frow = &fs[tid * CHUNK];
      const int jb = c * CHUNK;
      for (int q = 0; q < CHUNK; ++q) {
        float v = frow[q];
        if (v > h[0]) {                      // beats current k-th largest
          int p = 0;
          for (;;) {                         // sift-down (6 levels max)
            int ch = 2 * p + 1;
            if (ch >= TOPK) break;
            int c2 = ch + 1;
            if (c2 < TOPK && h[c2] < h[ch]) ch = c2;
            if (h[ch] >= v) break;
            h[p] = h[ch]; g[p] = g[ch]; p = ch;
          }
          h[p] = v; g[p] = jb + q;
        }
      }
    }
    __syncthreads();
  }

  // ---- sparse decode: out[t] = sum_k hv[t][k] * dec[hidx[t][k]] + b_dec ----
  const int dg = tid & 15;                   // dim group: dims dg + 16*i
  for (int tg = 0; tg < SUBT; ++tg) {
    const int t = tg * 16 + (tid >> 4);      // token within tile
    float a2[D_IN / 16];
#pragma unroll
    for (int i = 0; i < D_IN / 16; ++i) a2[i] = bdec[dg + 16 * i];

    for (int k = 0; k < TOPK; ++k) {
      float v = hv[t * TOPK + k];
      if (v > 0.0f) {                        // decoder rows live in 192MB L2
        const float* drow = &dec[(size_t)hidx[t * TOPK + k] * D_IN];
#pragma unroll
        for (int i = 0; i < D_IN / 16; ++i)
          a2[i] += v * drow[dg + 16 * i];
      }
    }
    float* orow = &out[(size_t)(tokBase + t) * D_IN];
#pragma unroll
    for (int i = 0; i < D_IN / 16; ++i) orow[dg + 16 * i] = a2[i];
  }
}

extern "C" void kernel_launch(void* const* d_in, const int* in_sizes, int n_in,
                              void* d_out, int out_size, void* d_ws, size_t ws_size,
                              hipStream_t stream) {
  const float* x    = (const float*)d_in[0];
  const float* W    = (const float*)d_in[1];
  const float* benc = (const float*)d_in[2];
  const float* dec  = (const float*)d_in[3];
  const float* bdec = (const float*)d_in[4];
  float* out  = (float*)d_out;
  __bf16* Wq  = (__bf16*)d_ws;

  const size_t needW = (size_t)D_SAE * D_IN * sizeof(__bf16);  // ~37.5 MB
  dim3 block(256);

  if (ws_size >= needW) {
    // pre-convert W_enc to swizzled bf16 fragments (one-time ~5us stream)
    hipLaunchKernelGGL(convert_W, dim3(NFRAG / 8), block, 0, stream, W, Wq);
    hipLaunchKernelGGL((sae_topk_fused<true>), dim3(N_TOK / TILE_M), block, 0,
                       stream, x, W, Wq, benc, dec, bdec, out);
  } else {
    hipLaunchKernelGGL((sae_topk_fused<false>), dim3(N_TOK / TILE_M), block, 0,
                       stream, x, W, Wq, benc, dec, bdec, out);
  }
}
